// CrossAttention_44693429682596
// MI455X (gfx1250) — matrix operand order repaired and checked by reading, not compiled, not run on hardware
//
#include <hip/hip_runtime.h>

#define BB 4
#define CC 256
#define CIc 128
#define NN 4096   // H*W = 64*64

typedef __attribute__((ext_vector_type(16))) _Float16 v16h;
typedef __attribute__((ext_vector_type(8)))  _Float16 v8h;
typedef __attribute__((ext_vector_type(4)))  _Float16 v4h;
typedef __attribute__((ext_vector_type(8)))  float    v8f;

typedef __attribute__((address_space(3))) uint4 lds_uint4;

__device__ __forceinline__ v8f wmma_f16(v16h a, v16h b, v8f c) {
  return __builtin_amdgcn_wmma_f32_16x16x32_f16(false, a, false, b, (short)0, c, false, false);
}

// CDNA5 async global->LDS copy (tracked by ASYNCcnt), 16B per lane.
__device__ __forceinline__ void async_copy_b128(lds_uint4* lds, const uint4* g) {
  asm volatile("global_load_async_to_lds_b128 %0, %1, off"
               :: "v"(lds), "v"(g) : "memory");
}
__device__ __forceinline__ void async_wait0() {
#if __has_builtin(__builtin_amdgcn_s_wait_asynccnt)
  __builtin_amdgcn_s_wait_asynccnt(0);
#else
  asm volatile("s_wait_asynccnt 0" ::: "memory");
#endif
}
__device__ __forceinline__ void async_wait8() {
#if __has_builtin(__builtin_amdgcn_s_wait_asynccnt)
  __builtin_amdgcn_s_wait_asynccnt(8);
#else
  asm volatile("s_wait_asynccnt 8" ::: "memory");
#endif
}

// A-fragment k index for vgpr-pair j (ISA 7.12.2, 16-bit A 16x32):
// lanes 0-15: K = {0..7, 16..23}; lanes 16-31: K = {8..15, 24..31}
__device__ __forceinline__ int a_kidx(int j, int half) {
  return ((j < 4) ? (2 * j) : (2 * (j - 4) + 16)) + half * 8;
}

// Fragment-order slot for element k (0..31): reader half h = (k>>3)&1,
// slot within the reader lane's 16 contiguous halfs. 4-aligned k -> 4-aligned
// slot, and 4 consecutive k stay in one (h, slot..slot+3) run.
__device__ __forceinline__ void a_slot(int k, int& h, int& slot) {
  h = (k >> 3) & 1;
  slot = (k < 16) ? (k - h * 8) : (k - 8 - h * 8);
}

// A(m,k) = S[m*ld + k]  (row-major 16x32 f16, global)
__device__ __forceinline__ v16h load_A(const _Float16* S, int ld) {
  int lane = threadIdx.x & 31, half = lane >> 4, r = lane & 15;
  const _Float16* row = S + r * ld;
  v16h a;
#pragma unroll
  for (int j = 0; j < 8; ++j) {
    int k = a_kidx(j, half);
    a[2 * j]     = row[k];
    a[2 * j + 1] = row[k + 1];
  }
  return a;
}

// A(m,k) from f32 source, convert to f16
__device__ __forceinline__ v16h load_A_f32(const float* S, int ld) {
  int lane = threadIdx.x & 31, half = lane >> 4, r = lane & 15;
  const float* row = S + r * ld;
  v16h a;
#pragma unroll
  for (int j = 0; j < 8; ++j) {
    int k = a_kidx(j, half);
    a[2 * j]     = (_Float16)row[k];
    a[2 * j + 1] = (_Float16)row[k + 1];
  }
  return a;
}

// B(k,n) = S[n*ld + k]: lane n=lane&15 reads K = {8h..8h+7, 16+8h..16+8h+7}
// (A-fragment order) -> two ds_load_b128.
__device__ __forceinline__ v16h load_B_T(const _Float16* S, int ld) {
  int lane = threadIdx.x & 31;
  int kh = (lane >> 4) * 8, n = lane & 15;
  const _Float16* row = S + n * ld + kh;
  v16h b;
#pragma unroll
  for (int j = 0; j < 16; ++j) b[j] = row[(j < 8) ? j : (j + 8)];
  return b;
}

// ---------------------------------------------------------------------------
// Kernel 1: fused 1x1-conv projections. grid = (N/64, B, 3)  block = 256
//   which=0: Q [B][N][CI] ; which=1: K [B][N][CI] ; which=2: V [B][CI][N]
// X tile staged in A-fragment order with 4-wide b64 scatter stores.
// ---------------------------------------------------------------------------
__global__ __launch_bounds__(256) void qkv_proj(
    const float* __restrict__ x, const float* __restrict__ rgbd,
    const float* __restrict__ qw, const float* __restrict__ qb,
    const float* __restrict__ kw, const float* __restrict__ kb,
    const float* __restrict__ vw, const float* __restrict__ vb,
    _Float16* __restrict__ Qb, _Float16* __restrict__ Kb,
    _Float16* __restrict__ Vb) {
  __shared__ __align__(32) _Float16 Xf[4 * 32 * 16];  // [tile][lane][slot]
  __shared__ __align__(32) _Float16 Wt[128 * 32];     // [o][c]

  int which = blockIdx.z;
  const float* X    = (which == 0) ? x  : rgbd;
  const float* W    = (which == 0) ? qw : (which == 1 ? kw : vw);
  const float* bias = (which == 0) ? qb : (which == 1 ? kb : vb);
  _Float16* Out     = (which == 0) ? Qb : (which == 1 ? Kb : Vb);

  int b  = blockIdx.y;
  int t0 = blockIdx.x * 64;
  int tid = threadIdx.x, lane = tid & 31, wave = tid >> 5;
  int half = lane >> 4, col = lane & 15;
  int wtok = wave & 3, wci = wave >> 2;
  const float* Xb = X + (size_t)b * CC * NN;

  v8f acc[4] = {};
  for (int c0 = 0; c0 < CC; c0 += 32) {
#pragma unroll
    for (int j = 0; j < 2; ++j) {           // X tile: 512 channel-quads
      int pe = tid + 256 * j;
      int c4 = (pe >> 6) * 4;               // 0,4,...,28
      int t  = pe & 63;
      const float* src = &Xb[(size_t)(c0 + c4) * NN + t0 + t];
      v4h pv;
      pv[0] = (_Float16)src[0];
      pv[1] = (_Float16)src[NN];
      pv[2] = (_Float16)src[2 * NN];
      pv[3] = (_Float16)src[3 * NN];
      int h, slot;
      a_slot(c4, h, slot);                  // slot 4-aligned -> b64 store
      *(v4h*)&Xf[((t >> 4) * 32 + h * 16 + (t & 15)) * 16 + slot] = pv;
    }
#pragma unroll
    for (int j = 0; j < 4; ++j) {           // W tile: 1024 quads, b128->b64
      int pe = tid + 256 * j;
      int o  = pe >> 3;                     // 0..127
      int c4 = (pe & 7) * 4;                // 0..28
      float4 wv = *(const float4*)&W[o * CC + c0 + c4];
      v4h hv;
      hv[0] = (_Float16)wv.x; hv[1] = (_Float16)wv.y;
      hv[2] = (_Float16)wv.z; hv[3] = (_Float16)wv.w;
      *(v4h*)&Wt[o * 32 + c4] = hv;
    }
    __syncthreads();

    v16h a = *(const v16h*)&Xf[(wtok * 32 + lane) * 16];  // 2x ds_load_b128
#pragma unroll
    for (int t = 0; t < 4; ++t) {           // B(k=c, n=o) from Wt[o][c]
      v16h bf = load_B_T(&Wt[(wci * 4 + t) * 512], 32);
      acc[t] = wmma_f16(a, bf, acc[t]);
    }
    __syncthreads();
  }

  if (which == 2) {                         // V: [B][CI][N], 8 tokens/b128
    _Float16* vout = Out + (size_t)b * CIc * NN + t0 + wtok * 16 + half * 8;
#pragma unroll
    for (int t = 0; t < 4; ++t) {
      int o = (wci * 4 + t) * 16 + col;
      float bv = bias[o];
      v8h pv;
#pragma unroll
      for (int i = 0; i < 8; ++i) pv[i] = (_Float16)(acc[t][i] + bv);
      *(v8h*)&vout[(size_t)o * NN] = pv;
    }
  } else {                                  // Q/K: [B][N][CI]
#pragma unroll
    for (int t = 0; t < 4; ++t) {
      int o = (wci * 4 + t) * 16 + col;
      float bv = bias[o];
#pragma unroll
      for (int i = 0; i < 8; ++i) {
        int token = t0 + wtok * 16 + i + half * 8;
        Out[((size_t)b * NN + token) * CIc + o] = (_Float16)(acc[t][i] + bv);
      }
    }
  }
}

// ---------------------------------------------------------------------------
// Kernel 2: flash attention, software-pipelined async DMA.
// grid = (N/64, B)  block = 128 (4 waves).  K staged [key][ci]; V [ci][key].
// ASYNCcnt completes in order: K(kb+1) prefetch overlaps softmax+PV,
// V(kb+1) prefetch overlaps the next S-compute. Single 40KB LDS footprint.
// ---------------------------------------------------------------------------
__global__ __launch_bounds__(128) void flash_attn(
    const _Float16* __restrict__ Qb, const _Float16* __restrict__ Kb,
    const _Float16* __restrict__ Vb, _Float16* __restrict__ Ob) {
  __shared__ __align__(16) _Float16 Kt[64 * CIc];    // [key][ci]   16 KB
  __shared__ __align__(16) _Float16 Vt[CIc * 64];    // [ci][key]   16 KB
  __shared__ __align__(32) _Float16 Pf[4 * 2 * 512]; // per-wave frag slabs 8KB

  int b = blockIdx.y;
  int tid = threadIdx.x, lane = tid & 31, wave = tid >> 5;
  int half = lane >> 4, col = lane & 15;
  int q0 = blockIdx.x * 64 + wave * 16;
  const int nb = NN / 64;

  lds_uint4* kd = (lds_uint4*)Kt;
  lds_uint4* vd = (lds_uint4*)Vt;
  auto issueK = [&](int blk) {
    const uint4* srcK = (const uint4*)(Kb + ((size_t)b * NN + blk * 64) * CIc);
#pragma unroll
    for (int j = 0; j < 8; ++j) {
      int e = tid + 128 * j;
      async_copy_b128(kd + e, srcK + e);                        // [key][ci]
    }
  };
  auto issueV = [&](int blk) {
    const uint4* srcV = (const uint4*)(Vb + (size_t)b * CIc * NN + blk * 64);
#pragma unroll
    for (int j = 0; j < 8; ++j) {
      int e = tid + 128 * j;
      async_copy_b128(vd + e, srcV + (e >> 3) * 512 + (e & 7)); // [ci][key]
    }
  };

  const _Float16* Qrow = Qb + ((size_t)b * NN + q0) * CIc;
  v16h qf[4];
#pragma unroll
  for (int kk = 0; kk < 4; ++kk) qf[kk] = load_A(Qrow + kk * 32, CIc);

  v8f oacc[8] = {};
  float m[8], l[8];
#pragma unroll
  for (int i = 0; i < 8; ++i) { m[i] = -1e30f; l[i] = 0.f; }
  _Float16* Pw = &Pf[wave * 1024];

  // prologue: K(0) resident, V(0) in flight
  issueK(0);
  async_wait0();
  __syncthreads();
  issueV(0);

  for (int kb = 0; kb < nb; ++kb) {
    bool more = (kb + 1) < nb;

    // S = Q * K^T  (16 x 64, f32) from resident Kt
    v8f s[4] = {};
#pragma unroll
    for (int ks = 0; ks < 4; ++ks) {
#pragma unroll
      for (int kk = 0; kk < 4; ++kk) {       // B(k=ci, n=key) from Kt[key][ci]
        v16h bf = load_B_T(&Kt[(ks * 16) * CIc + kk * 32], CIc);
        s[ks] = wmma_f16(qf[kk], bf, s[ks]);
      }
    }
    __syncthreads();                         // all waves done reading Kt
    if (more) issueK(kb + 1);                // overlaps softmax + PV

    // online softmax; row i lives across the 16 lanes of this half-wave
#pragma unroll
    for (int i = 0; i < 8; ++i) {
      float rmax = s[0][i];
#pragma unroll
      for (int ks = 1; ks < 4; ++ks) rmax = fmaxf(rmax, s[ks][i]);
#pragma unroll
      for (int d = 1; d < 16; d <<= 1) rmax = fmaxf(rmax, __shfl_xor(rmax, d, 32));
      float newm = fmaxf(m[i], rmax);
      float corr = __expf(m[i] - newm);
      float rsum = 0.f;
#pragma unroll
      for (int ks = 0; ks < 4; ++ks) {
        float p = __expf(s[ks][i] - newm);
        s[ks][i] = p;
        rsum += p;
      }
#pragma unroll
      for (int d = 1; d < 16; d <<= 1) rsum += __shfl_xor(rsum, d, 32);
      l[i] = l[i] * corr + rsum;
      m[i] = newm;
#pragma unroll
      for (int t = 0; t < 8; ++t) oacc[t][i] *= corr;
    }

    if (more) async_wait8();                 // oldest 8 (= V(kb)) retired
    else      async_wait0();
    __syncthreads();                         // Vt visible to all waves

    // write P directly in A-fragment order (per-wave slab; DS is in-order
    // within a wave so no barrier needed between write and read)
#pragma unroll
    for (int ks = 0; ks < 4; ++ks)
#pragma unroll
      for (int i = 0; i < 8; ++i) {
        int kg = ks * 16 + col;              // global key column 0..63
        int f = kg >> 5, k = kg & 31;
        int h, slot;
        a_slot(k, h, slot);
        Pw[f * 512 + (h * 16 + (i + half * 8)) * 16 + slot] = (_Float16)s[ks][i];
      }
    v16h pf0 = *(const v16h*)&Pw[lane * 16];        // keys 0..31
    v16h pf1 = *(const v16h*)&Pw[512 + lane * 16];  // keys 32..63

    // O += P * V ;  B(k=key, n=ci) from Vt[ci][key] -> contiguous reads
#pragma unroll
    for (int t = 0; t < 8; ++t) {
      v16h b0 = load_B_T(&Vt[(t * 16) * 64 + 0], 64);
      oacc[t] = wmma_f16(pf0, b0, oacc[t]);
      v16h b1 = load_B_T(&Vt[(t * 16) * 64 + 32], 64);
      oacc[t] = wmma_f16(pf1, b1, oacc[t]);
    }

    async_wait0();                           // K(kb+1) complete
    __syncthreads();                         // Vt reads done + Kt visible
    if (more) issueV(kb + 1);                // overlaps next S-compute
  }

#pragma unroll
  for (int i = 0; i < 8; ++i) l[i] = __frcp_rn(l[i]);
#pragma unroll
  for (int t = 0; t < 8; ++t)
#pragma unroll
    for (int i = 0; i < 8; ++i) {
      int token = q0 + i + half * 8;
      Ob[((size_t)b * NN + token) * CIc + t * 16 + col] =
          (_Float16)(oacc[t][i] * l[i]);
    }
}

// ---------------------------------------------------------------------------
// Kernel 3: output projection + bias + residual.
// grid = (N/16, C/128, B), block = 256 (8 waves; wave w -> 16 output channels)
// ---------------------------------------------------------------------------
__global__ __launch_bounds__(256) void out_proj(
    const _Float16* __restrict__ Ob, const float* __restrict__ Wo,
    const float* __restrict__ bo, const float* __restrict__ rgbd,
    float* __restrict__ out) {
  __shared__ __align__(16) _Float16 Ot[16 * CIc];   // [token][ci], 4 KB

  int b = blockIdx.z;
  int n0 = blockIdx.x * 16;
  int tid = threadIdx.x, lane = tid & 31, wave = tid >> 5;
  int half = lane >> 4, col = lane & 15;
  int cobase = blockIdx.y * 128 + wave * 16;

  ((uint4*)Ot)[tid] = ((const uint4*)(Ob + ((size_t)b * NN + n0) * CIc))[tid];
  __syncthreads();

  v8f acc = {};
#pragma unroll
  for (int kk = 0; kk < 4; ++kk) {
    v16h af = load_A_f32(Wo + (size_t)cobase * CIc + kk * 32, CIc); // A(co,ci)
    v16h bf = load_B_T(&Ot[kk * 32], CIc);                          // B(ci,tok)
    acc = wmma_f16(af, bf, acc);
  }

  int token = n0 + col;
#pragma unroll
  for (int i = 0; i < 8; ++i) {
    int co = cobase + i + half * 8;
    size_t idx = ((size_t)b * CC + co) * NN + token;
    out[idx] = rgbd[idx] + bo[co] + acc[i];
  }
}

extern "C" void kernel_launch(void* const* d_in, const int* in_sizes, int n_in,
                              void* d_out, int out_size, void* d_ws, size_t ws_size,
                              hipStream_t stream) {
  const float* rgbd = (const float*)d_in[0];
  const float* x    = (const float*)d_in[1];
  const float* qw   = (const float*)d_in[2];
  const float* qb   = (const float*)d_in[3];
  const float* kw   = (const float*)d_in[4];
  const float* kb   = (const float*)d_in[5];
  const float* vw   = (const float*)d_in[6];
  const float* vb   = (const float*)d_in[7];
  const float* ow   = (const float*)d_in[8];
  const float* ob   = (const float*)d_in[9];
  float* out = (float*)d_out;

  size_t per = (size_t)BB * NN * CIc;       // 4 MB f16 each
  _Float16* Qb = (_Float16*)d_ws;
  _Float16* Kb = Qb + per;
  _Float16* Vb = Kb + per;                  // V stored [B][CI][N]
  _Float16* Ob = Vb + per;

  qkv_proj<<<dim3(NN / 64, BB, 3), 256, 0, stream>>>(x, rgbd, qw, qb, kw, kb,
                                                     vw, vb, Qb, Kb, Vb);
  flash_attn<<<dim3(NN / 64, BB), 128, 0, stream>>>(Qb, Kb, Vb, Ob);
  out_proj<<<dim3(NN / 16, CC / 128, BB), 256, 0, stream>>>(Ob, ow, ob, rgbd,
                                                            out);
}
